// Zamba2MambaDecoderLayer_30382598652294
// MI455X (gfx1250) — compile-verified, matrix operand order checked by reading, not run on hardware
//
#include <hip/hip_runtime.h>
#include <hip/hip_bf16.h>
#include <math.h>

// ---------------- problem constants ----------------
#define H_SIZE   2560
#define INTER    5120
#define D_STATE  64
#define D_CONV   4
#define NHEADS   64
#define HEAD_DIM 80
#define CHUNK    256
#define NCHUNK   8
#define BATCH    2
#define SEQLEN   2048
#define ROWS     (BATCH * SEQLEN)          // 4096
#define CONV_DIM (INTER + 2 * D_STATE)     // 5248
#define PROJ_DIM (2 * INTER + 2 * D_STATE + NHEADS) // 10432
#define DT_OFF   (INTER + CONV_DIM)        // 10368
#define EPS      1e-5f
#define UNITS    (BATCH * NCHUNK * NHEADS) // 1024

typedef __bf16 bf16_t;
typedef __attribute__((ext_vector_type(16))) __bf16 v16bf;
typedef __attribute__((ext_vector_type(8)))  float  v8f;

// WMMA bf16 fragment load from a row-major [N x K] (or [M x K]) matrix.
// 16-bit A layout: lane = (half)*16 + row; elements 0..7 -> k = kb..kb+7,
// elements 8..15 -> k = kb+16..kb+23, kb = (lane>>4)*8.
__device__ __forceinline__ v16bf ldfrag_g(const bf16_t* p, int ld, int lane) {
  int r = lane & 15, kb = (lane >> 4) << 3;
  const bf16_t* q = p + (size_t)r * ld + kb;
  v16bf f;
#pragma unroll
  for (int i = 0; i < 8; ++i) { f[i] = q[i]; f[i + 8] = q[i + 16]; }
  return f;
}

__device__ __forceinline__ float silu_f(float x) { return x / (1.0f + expf(-x)); }

// ---------------- elementwise / prep kernels ----------------
__global__ void f32_to_bf16_kernel(const float* __restrict__ in, bf16_t* __restrict__ out, long n) {
  long i = (long)blockIdx.x * blockDim.x + threadIdx.x;
  long stride = (long)gridDim.x * blockDim.x;
  for (; i < n; i += stride) out[i] = (bf16_t)in[i];
}

__global__ void rmsnorm1_kernel(const float* __restrict__ x, const float* __restrict__ w,
                                bf16_t* __restrict__ out) {
  __shared__ float red[256];
  size_t r = blockIdx.x;
  const float* xr = x + r * H_SIZE;
  float s = 0.f;
  for (int i = threadIdx.x; i < H_SIZE; i += 256) { float v = xr[i]; s += v * v; }
  red[threadIdx.x] = s; __syncthreads();
  for (int off = 128; off > 0; off >>= 1) {
    if ((int)threadIdx.x < off) red[threadIdx.x] += red[threadIdx.x + off];
    __syncthreads();
  }
  float sc = rsqrtf(red[0] / (float)H_SIZE + EPS);
  for (int i = threadIdx.x; i < H_SIZE; i += 256)
    out[r * H_SIZE + i] = (bf16_t)(xr[i] * sc * w[i]);
}

// ---------------- WMMA GEMM: C[M,N] = A[M,K] * W[N,K]^T (+ addsrc) ----------------
// Each wave owns a 16(M) x 32(N) output (two accumulators sharing one A frag).
// 8 waves/block arranged 4(M) x 2(N) -> 64x64 block tile. 2-stage register
// pipeline: next k-step's fragments are loaded before the current wmma pair,
// so the matrix pipe overlaps outstanding global loads (partial loadcnt waits).
__global__ void gemm_bf16_kernel(const bf16_t* __restrict__ A, const bf16_t* __restrict__ W,
                                 const float* __restrict__ addsrc, float* __restrict__ C,
                                 int M, int N, int K) {
  int lane = threadIdx.x & 31;
  int w = threadIdx.x >> 5;                       // 8 waves
  int m0 = (blockIdx.y * 4 + (w >> 1)) * 16;      // 64 rows / block
  int n0 = (blockIdx.x * 2 + (w & 1)) * 32;       // 64 cols / block
  const bf16_t* Ap = A + (size_t)m0 * K;
  const bf16_t* W0 = W + (size_t)n0 * K;
  const bf16_t* W1 = W + (size_t)(n0 + 16) * K;
  v8f acc0 = {}, acc1 = {};
  v16bf a  = ldfrag_g(Ap, K, lane);
  v16bf b0 = ldfrag_g(W0, K, lane);
  v16bf b1 = ldfrag_g(W1, K, lane);
  for (int k = 32; k < K; k += 32) {
    if (k + 224 < K) {
      __builtin_prefetch(Ap + ((size_t)(lane & 15)) * K + k + 224, 0, 0);
      __builtin_prefetch(W0 + ((size_t)(lane & 15)) * K + k + 224, 0, 0);
      __builtin_prefetch(W1 + ((size_t)(lane & 15)) * K + k + 224, 0, 0);
    }
    v16bf an  = ldfrag_g(Ap + k, K, lane);
    v16bf bn0 = ldfrag_g(W0 + k, K, lane);
    v16bf bn1 = ldfrag_g(W1 + k, K, lane);
    acc0 = __builtin_amdgcn_wmma_f32_16x16x32_bf16(false, a, false, b0, (short)0, acc0, false, false);
    acc1 = __builtin_amdgcn_wmma_f32_16x16x32_bf16(false, a, false, b1, (short)0, acc1, false, false);
    a = an; b0 = bn0; b1 = bn1;
  }
  acc0 = __builtin_amdgcn_wmma_f32_16x16x32_bf16(false, a, false, b0, (short)0, acc0, false, false);
  acc1 = __builtin_amdgcn_wmma_f32_16x16x32_bf16(false, a, false, b1, (short)0, acc1, false, false);

  int n = n0 + (lane & 15);
  int mb = m0 + ((lane >> 4) << 3);
#pragma unroll
  for (int r = 0; r < 8; ++r) {
    size_t i0 = (size_t)(mb + r) * N + n;
    size_t i1 = i0 + 16;
    float v0 = acc0[r], v1 = acc1[r];
    if (addsrc) { v0 += addsrc[i0]; v1 += addsrc[i1]; }
    C[i0] = v0;
    C[i1] = v1;
  }
}

// ---------------- depthwise causal conv (d=4) + SiLU ----------------
__global__ void conv_silu_kernel(const float* __restrict__ proj, const float* __restrict__ cw,
                                 const float* __restrict__ cb, float* __restrict__ xBC) {
  long idx = (long)blockIdx.x * 256 + threadIdx.x;
  if (idx >= (long)ROWS * CONV_DIM) return;
  int c = (int)(idx % CONV_DIM);
  int r = (int)(idx / CONV_DIM);
  int b = r / SEQLEN, l = r % SEQLEN;
  float s = cb[c];
#pragma unroll
  for (int k = 0; k < D_CONV; ++k) {
    int pos = l - (D_CONV - 1) + k;
    if (pos >= 0)
      s += proj[(size_t)(b * SEQLEN + pos) * PROJ_DIM + INTER + c] * cw[c * D_CONV + k];
  }
  xBC[(size_t)r * CONV_DIM + c] = silu_f(s);
}

__global__ void dt_kernel(const float* __restrict__ proj, const float* __restrict__ dt_bias,
                          float* __restrict__ dt) {
  int idx = blockIdx.x * 256 + threadIdx.x;   // ROWS*NHEADS
  int r = idx >> 6, h = idx & 63;
  float v = proj[(size_t)r * PROJ_DIM + DT_OFF + h] + dt_bias[h];
  dt[idx] = (v > 20.f) ? v : log1pf(expf(v));
}

// ---------------- per-chunk inclusive cumsum of a = dt * A ----------------
// block = unit = ((b*8+c)*64+h); acs[unit*256+s], asum[unit]
__global__ void acs_kernel(const float* __restrict__ dt, const float* __restrict__ A_log,
                           float* __restrict__ acs, float* __restrict__ asum) {
  __shared__ float buf[2][256];
  int h = blockIdx.x & 63, c = (blockIdx.x >> 6) & 7, b = blockIdx.x >> 9;
  int s = threadIdx.x;
  float Ah = -expf(A_log[h]);
  float a = dt[(size_t)(b * SEQLEN + c * CHUNK + s) * NHEADS + h] * Ah;
  buf[0][s] = a; __syncthreads();
  int src = 0;
  for (int off = 1; off < 256; off <<= 1) {
    float v = buf[src][s];
    if (s >= off) v += buf[src][s - off];
    buf[src ^ 1][s] = v;
    src ^= 1; __syncthreads();
  }
  float cs = buf[src][s];
  acs[(size_t)blockIdx.x * 256 + s] = cs;
  if (s == 255) asum[blockIdx.x] = cs;
}

// ---------------- chunk states: S[p,n] = sum_s exp(aend-acs[s])*dt[s]*x[s,p]*B[s,n] ----------------
__global__ void states_kernel(const float* __restrict__ xBC, const float* __restrict__ dt,
                              const float* __restrict__ acs, const float* __restrict__ asum,
                              float* __restrict__ states) {
  __shared__ float sxw[32][80];
  __shared__ float sb[32][64];
  int h = blockIdx.x & 63, c = (blockIdx.x >> 6) & 7, b = blockIdx.x >> 9;
  int rowb = b * SEQLEN + c * CHUNK;
  float aend = asum[blockIdx.x];
  float acc[20];
#pragma unroll
  for (int j = 0; j < 20; ++j) acc[j] = 0.f;
  for (int st = 0; st < 256; st += 32) {
    for (int i = threadIdx.x; i < 32 * 80; i += 256) {
      int s = st + i / 80, p = i % 80;
      float wgt = expf(aend - acs[(size_t)blockIdx.x * 256 + s]) *
                  dt[(size_t)(rowb + s) * NHEADS + h];
      sxw[i / 80][p] = xBC[(size_t)(rowb + s) * CONV_DIM + h * HEAD_DIM + p] * wgt;
    }
    for (int i = threadIdx.x; i < 32 * 64; i += 256) {
      int s = st + i / 64, n = i % 64;
      sb[i / 64][n] = xBC[(size_t)(rowb + s) * CONV_DIM + INTER + n];
    }
    __syncthreads();
#pragma unroll
    for (int j = 0; j < 20; ++j) {
      int idx = threadIdx.x + j * 256;       // 0..5119
      int p = idx >> 6, n = idx & 63;
      float s = acc[j];
#pragma unroll 8
      for (int ss = 0; ss < 32; ++ss) s += sxw[ss][p] * sb[ss][n];
      acc[j] = s;
    }
    __syncthreads();
  }
#pragma unroll
  for (int j = 0; j < 20; ++j)
    states[(size_t)blockIdx.x * 5120 + threadIdx.x + j * 256] = acc[j];
}

// ---------------- inter-chunk scan: prev_{c+1} = exp(asum_c)*prev_c + S_c ----------------
__global__ void scan_kernel(const float* __restrict__ states, const float* __restrict__ asum,
                            float* __restrict__ prevs) {
  int h = blockIdx.x & 63, b = blockIdx.x >> 6;
  float prev[20];
#pragma unroll
  for (int j = 0; j < 20; ++j) prev[j] = 0.f;
  for (int c = 0; c < NCHUNK; ++c) {
    int unit = (b * NCHUNK + c) * NHEADS + h;
    float dec = expf(asum[unit]);
#pragma unroll
    for (int j = 0; j < 20; ++j) {
      int idx = threadIdx.x + j * 256;
      prevs[(size_t)unit * 5120 + idx] = prev[j];
      prev[j] = prev[j] * dec + states[(size_t)unit * 5120 + idx];
    }
  }
}

// ---------------- G = (C * B^T) .* L  (masked decay), bf16 out ----------------
__global__ void gtile_kernel(const float* __restrict__ xBC, const float* __restrict__ acs,
                             bf16_t* __restrict__ Gbuf) {
  __shared__ bf16_t sB[256][64];
  __shared__ float sA[256];
  int c = (blockIdx.x >> 6) & 7, b = blockIdx.x >> 9;
  int rowb = b * SEQLEN + c * CHUNK;
  int lane = threadIdx.x & 31, w = threadIdx.x >> 5;
  for (int i = threadIdx.x; i < 256 * 64; i += 256) {
    int s = i >> 6, n = i & 63;
    sB[s][n] = (bf16_t)xBC[(size_t)(rowb + s) * CONV_DIM + INTER + n];
  }
  sA[threadIdx.x] = acs[(size_t)blockIdx.x * 256 + threadIdx.x];
  __syncthreads();
  int r = lane & 15, kb = (lane >> 4) << 3;
  for (int strip = w; strip < 16; strip += 8) {
    int l0 = strip * 16;
    v16bf a0, a1;
    {
      const float* q = xBC + (size_t)(rowb + l0 + r) * CONV_DIM + INTER + D_STATE;
#pragma unroll
      for (int i = 0; i < 8; ++i) {
        a0[i] = (bf16_t)q[kb + i];      a0[i + 8] = (bf16_t)q[kb + 16 + i];
        a1[i] = (bf16_t)q[32 + kb + i]; a1[i + 8] = (bf16_t)q[32 + kb + 16 + i];
      }
    }
    for (int st = 0; st < 16; ++st) {
      v8f g = {};
      if (st <= strip) {
        v16bf b0, b1;
#pragma unroll
        for (int i = 0; i < 8; ++i) {
          b0[i] = sB[st * 16 + r][kb + i];      b0[i + 8] = sB[st * 16 + r][kb + 16 + i];
          b1[i] = sB[st * 16 + r][32 + kb + i]; b1[i + 8] = sB[st * 16 + r][32 + kb + 16 + i];
        }
        g = __builtin_amdgcn_wmma_f32_16x16x32_bf16(false, a0, false, b0, (short)0, g, false, false);
        g = __builtin_amdgcn_wmma_f32_16x16x32_bf16(false, a1, false, b1, (short)0, g, false, false);
      }
      int sl = st * 16 + (lane & 15);
      int mb = l0 + ((lane >> 4) << 3);
      float as = sA[sl];
#pragma unroll
      for (int rr = 0; rr < 8; ++rr) {
        int ll = mb + rr;
        float f = (ll >= sl) ? expf(sA[ll] - as) : 0.0f;
        Gbuf[(size_t)blockIdx.x * 65536 + (size_t)ll * 256 + sl] = (bf16_t)(g[rr] * f);
      }
    }
  }
}

// ---------------- Y = G*X + exp(acs)*(C*prev^T) + D*x ----------------
__global__ void ydiag_kernel(const float* __restrict__ xBC, const float* __restrict__ dt,
                             const float* __restrict__ acs, const float* __restrict__ prevs,
                             const bf16_t* __restrict__ Gbuf, const float* __restrict__ Dvec,
                             float* __restrict__ y) {
  __shared__ bf16_t sX[256][80];
  __shared__ bf16_t sP[80][64];
  __shared__ float sA[256];
  int h = blockIdx.x & 63, c = (blockIdx.x >> 6) & 7, b = blockIdx.x >> 9;
  int rowb = b * SEQLEN + c * CHUNK;
  int lane = threadIdx.x & 31, w = threadIdx.x >> 5;
  for (int i = threadIdx.x; i < 256 * 80; i += 256) {
    int s = i / 80, p = i % 80;
    sX[s][p] = (bf16_t)(xBC[(size_t)(rowb + s) * CONV_DIM + h * HEAD_DIM + p] *
                        dt[(size_t)(rowb + s) * NHEADS + h]);
  }
  for (int i = threadIdx.x; i < 80 * 64; i += 256)
    sP[i >> 6][i & 63] = (bf16_t)prevs[(size_t)blockIdx.x * 5120 + i];
  sA[threadIdx.x] = acs[(size_t)blockIdx.x * 256 + threadIdx.x];
  __syncthreads();
  float Dh = Dvec[h];
  int r = lane & 15, kb = (lane >> 4) << 3;
  for (int strip = w; strip < 16; strip += 8) {
    int l0 = strip * 16;
    v8f acc[5], off[5];
#pragma unroll
    for (int t = 0; t < 5; ++t) { acc[t] = (v8f){}; off[t] = (v8f){}; }
    const bf16_t* Gp = Gbuf + (size_t)blockIdx.x * 65536 + (size_t)l0 * 256;
    for (int k = 0; k < 256; k += 32) {
      v16bf a = ldfrag_g(Gp + k, 256, lane);
#pragma unroll
      for (int t = 0; t < 5; ++t) {
        v16bf bb;
#pragma unroll
        for (int i = 0; i < 8; ++i) {
          bb[i] = sX[k + kb + i][t * 16 + r];
          bb[i + 8] = sX[k + kb + 16 + i][t * 16 + r];
        }
        acc[t] = __builtin_amdgcn_wmma_f32_16x16x32_bf16(false, a, false, bb, (short)0, acc[t], false, false);
      }
    }
    v16bf ca0, ca1;
    {
      const float* q = xBC + (size_t)(rowb + l0 + r) * CONV_DIM + INTER + D_STATE;
#pragma unroll
      for (int i = 0; i < 8; ++i) {
        ca0[i] = (bf16_t)q[kb + i];      ca0[i + 8] = (bf16_t)q[kb + 16 + i];
        ca1[i] = (bf16_t)q[32 + kb + i]; ca1[i + 8] = (bf16_t)q[32 + kb + 16 + i];
      }
    }
#pragma unroll
    for (int t = 0; t < 5; ++t) {
      v16bf b0, b1;
#pragma unroll
      for (int i = 0; i < 8; ++i) {
        b0[i] = sP[t * 16 + r][kb + i];      b0[i + 8] = sP[t * 16 + r][kb + 16 + i];
        b1[i] = sP[t * 16 + r][32 + kb + i]; b1[i + 8] = sP[t * 16 + r][32 + kb + 16 + i];
      }
      off[t] = __builtin_amdgcn_wmma_f32_16x16x32_bf16(false, ca0, false, b0, (short)0, off[t], false, false);
      off[t] = __builtin_amdgcn_wmma_f32_16x16x32_bf16(false, ca1, false, b1, (short)0, off[t], false, false);
    }
    int pl = lane & 15, mb = l0 + ((lane >> 4) << 3);
#pragma unroll
    for (int rr = 0; rr < 8; ++rr) {
      int l = mb + rr;
      float sc = expf(sA[l]);
      size_t row = (size_t)(rowb + l);
#pragma unroll
      for (int t = 0; t < 5; ++t) {
        int p = t * 16 + pl;
        float xv = xBC[row * CONV_DIM + h * HEAD_DIM + p];
        y[row * INTER + h * HEAD_DIM + p] = acc[t][rr] + off[t][rr] * sc + Dh * xv;
      }
    }
  }
}

// ---------------- gate + rmsnorm2 -> bf16 ----------------
__global__ void gate_rmsnorm_kernel(const float* __restrict__ y, const float* __restrict__ proj,
                                    const float* __restrict__ nw, bf16_t* __restrict__ out) {
  __shared__ float red[256];
  size_t rr = blockIdx.x;
  const float* yr = y + rr * INTER;
  const float* zr = proj + rr * PROJ_DIM;
  float s = 0.f;
  for (int i = threadIdx.x; i < INTER; i += 256) {
    float v = yr[i] * silu_f(zr[i]);
    s += v * v;
  }
  red[threadIdx.x] = s; __syncthreads();
  for (int off = 128; off > 0; off >>= 1) {
    if ((int)threadIdx.x < off) red[threadIdx.x] += red[threadIdx.x + off];
    __syncthreads();
  }
  float sc = rsqrtf(red[0] / (float)INTER + EPS);
  for (int i = threadIdx.x; i < INTER; i += 256) {
    float v = yr[i] * silu_f(zr[i]);
    out[rr * INTER + i] = (bf16_t)(v * sc * nw[i]);
  }
}

// ---------------- launch ----------------
extern "C" void kernel_launch(void* const* d_in, const int* in_sizes, int n_in,
                              void* d_out, int out_size, void* d_ws, size_t ws_size,
                              hipStream_t stream) {
  const float* hidden   = (const float*)d_in[0];
  const float* ln_w     = (const float*)d_in[1];
  const float* in_w     = (const float*)d_in[2];
  const float* conv_w   = (const float*)d_in[3];
  const float* conv_b   = (const float*)d_in[4];
  const float* A_log    = (const float*)d_in[5];
  const float* Dvec     = (const float*)d_in[6];
  const float* dt_bias  = (const float*)d_in[7];
  const float* norm_w   = (const float*)d_in[8];
  const float* out_w    = (const float*)d_in[9];
  float* out = (float*)d_out;

  uint8_t* base = (uint8_t*)d_ws;
  size_t off = 0;
  auto alloc = [&](size_t bytes) { size_t o = off; off += (bytes + 255) & ~(size_t)255; return o; };

  bf16_t* Win   = (bf16_t*)(base + alloc((size_t)PROJ_DIM * H_SIZE * 2));
  bf16_t* Wout  = (bf16_t*)(base + alloc((size_t)H_SIZE * INTER * 2));
  bf16_t* xbf   = (bf16_t*)(base + alloc((size_t)ROWS * H_SIZE * 2));
  float*  proj  = (float*) (base + alloc((size_t)ROWS * PROJ_DIM * 4));
  float*  xBC   = (float*) (base + alloc((size_t)ROWS * CONV_DIM * 4));
  float*  dtb   = (float*) (base + alloc((size_t)ROWS * NHEADS * 4));
  float*  acs   = (float*) (base + alloc((size_t)UNITS * 256 * 4));
  float*  asum  = (float*) (base + alloc((size_t)UNITS * 4));
  float*  states= (float*) (base + alloc((size_t)UNITS * 5120 * 4));
  float*  prevs = (float*) (base + alloc((size_t)UNITS * 5120 * 4));
  bf16_t* Gbuf  = (bf16_t*)(base + alloc((size_t)UNITS * 65536 * 2));
  float*  ybuf  = (float*) (base + alloc((size_t)ROWS * INTER * 4));
  bf16_t* yg    = (bf16_t*)(base + alloc((size_t)ROWS * INTER * 2));

  // 1. weights -> bf16
  {
    long n1 = (long)PROJ_DIM * H_SIZE, n2 = (long)H_SIZE * INTER;
    f32_to_bf16_kernel<<<(int)((n1 + 255) / 256), 256, 0, stream>>>(in_w, Win, n1);
    f32_to_bf16_kernel<<<(int)((n2 + 255) / 256), 256, 0, stream>>>(out_w, Wout, n2);
  }
  // 2. rmsnorm1 -> bf16
  rmsnorm1_kernel<<<ROWS, 256, 0, stream>>>(hidden, ln_w, xbf);
  // 3. in_proj GEMM (WMMA): 64x64 block tiles
  gemm_bf16_kernel<<<dim3(PROJ_DIM / 64, ROWS / 64), 256, 0, stream>>>(
      xbf, Win, nullptr, proj, ROWS, PROJ_DIM, H_SIZE);
  // 4. conv + silu ; dt softplus
  {
    long n = (long)ROWS * CONV_DIM;
    conv_silu_kernel<<<(int)((n + 255) / 256), 256, 0, stream>>>(proj, conv_w, conv_b, xBC);
  }
  dt_kernel<<<ROWS * NHEADS / 256, 256, 0, stream>>>(proj, dt_bias, dtb);
  // 5. SSD
  acs_kernel<<<UNITS, 256, 0, stream>>>(dtb, A_log, acs, asum);
  states_kernel<<<UNITS, 256, 0, stream>>>(xBC, dtb, acs, asum, states);
  scan_kernel<<<BATCH * NHEADS, 256, 0, stream>>>(states, asum, prevs);
  gtile_kernel<<<UNITS, 256, 0, stream>>>(xBC, acs, Gbuf);
  ydiag_kernel<<<UNITS, 256, 0, stream>>>(xBC, dtb, acs, prevs, Gbuf, Dvec, ybuf);
  // 6. gate + rmsnorm2 -> bf16
  gate_rmsnorm_kernel<<<ROWS, 256, 0, stream>>>(ybuf, proj, norm_w, yg);
  // 7. out_proj GEMM (WMMA) + residual
  gemm_bf16_kernel<<<dim3(H_SIZE / 64, ROWS / 64), 256, 0, stream>>>(
      yg, Wout, hidden, out, ROWS, H_SIZE, INTER);
  (void)in_sizes; (void)n_in; (void)out_size; (void)ws_size;
}
